// GraphGP_91285234909685
// MI455X (gfx1250) — matrix operand order repaired and checked by reading, not compiled, not run on hardware
//
#include <hip/hip_runtime.h>
#include <hip/hip_bf16.h>

typedef __attribute__((ext_vector_type(16))) _Float16 v16h;
typedef __attribute__((ext_vector_type(8)))  _Float16 v8h;
typedef __attribute__((ext_vector_type(8)))  float    v8f;

union F16x16 { v16h v; v8h h[2]; };

static constexpr int NN_ = 4096;   // total nodes
static constexpr int SS_ = 2048;   // num_source
static constexpr int DD_ = 128;    // feature dim

__device__ __forceinline__ float softplusf(float x) {
  return fmaxf(x, 0.0f) + log1pf(__expf(-fabsf(x)));
}

// ---------------------------------------------------------------------------
// prep: xh = f16(feat/ls), norms[i] = sum(x^2)
// ---------------------------------------------------------------------------
__global__ void prep_x_kernel(const float* __restrict__ feat,
                              const float* __restrict__ raw_ls,
                              _Float16* __restrict__ xh,
                              float* __restrict__ norms) {
  __shared__ float red[128];
  const int i = blockIdx.x;
  const int d = threadIdx.x;
  const float ls = softplusf(raw_ls[0]);
  float x = feat[(size_t)i * DD_ + d] / ls;
  xh[(size_t)i * DD_ + d] = (_Float16)x;
  red[d] = x * x;
  __syncthreads();
  for (int s = 64; s > 0; s >>= 1) {
    if (d < s) red[d] += red[d + s];
    __syncthreads();
  }
  if (d == 0) norms[i] = red[0];
}

// ---------------------------------------------------------------------------
// pack adjacency: Ah = f16(adj * block_mask) for both adjacency matrices
// ---------------------------------------------------------------------------
__global__ void pack_adj_kernel(const float* __restrict__ adj,
                                _Float16* __restrict__ Ah) {
  size_t idx = (size_t)blockIdx.x * 256 + threadIdx.x;
  size_t nn = (size_t)NN_ * NN_;
  size_t e = idx % nn;
  int i = (int)(e / NN_);
  int j = (int)(e % NN_);
  float m = ((i < SS_) == (j < SS_)) ? 1.0f : 0.0f;
  Ah[idx] = (_Float16)(adj[idx] * m);
}

// ---------------------------------------------------------------------------
// Tiled WMMA f16 GEMM, 128x128 block tile, BK=32, 8 waves (wave32).
// MODE 0: NT, RBF epilogue  -> f16 out (C0)
// MODE 1: NN, plain         -> f16 out (A @ K)
// MODE 2: NT, affine accum  -> f32 out ((sb^2 + sw^2*T)*sa*ta, += if accumulate)
// ---------------------------------------------------------------------------
#define BM 128
#define BN 128
#define BK 32
#define LDST 40   // padded LDS row stride in halves (avoids bank conflicts)

template <int MODE>
__global__ __launch_bounds__(256) void gemm_kernel(
    const _Float16* __restrict__ A, const _Float16* __restrict__ B,
    void* __restrict__ Cout, const float* __restrict__ norms,
    const float* __restrict__ raw_sb, const float* __restrict__ raw_sw,
    const float* __restrict__ raw_sa, const float* __restrict__ raw_ta,
    int accumulate, int Kdim, int lda, int ldb, int ldc) {
  __shared__ __align__(16) _Float16 lA[BM * LDST];
  __shared__ __align__(16) _Float16 lB[BN * LDST];

  const int tid  = threadIdx.x;
  const int lane = tid & 31;
  const int wave = tid >> 5;
  const int wm   = wave >> 2;  // 0..1
  const int wn   = wave & 3;   // 0..3
  const int m0   = blockIdx.y * BM;
  const int n0   = blockIdx.x * BN;

  const int row_a = lane & 15;
  const int hi    = lane >> 4;  // 0 or 1

  v8f acc[4][2];
#pragma unroll
  for (int i = 0; i < 4; ++i)
#pragma unroll
    for (int j = 0; j < 2; ++j)
#pragma unroll
      for (int e = 0; e < 8; ++e) acc[i][j][e] = 0.0f;

  for (int kb = 0; kb < Kdim; kb += BK) {
    __syncthreads();
    // ---- stage A tile: lA[m][k] = A[m0+m][kb+k]
#pragma unroll
    for (int it = 0; it < 2; ++it) {
      int c  = tid + it * 256;     // 512 chunks of 8 halves
      int r  = c >> 2;
      int kc = (c & 3) * 8;
      *(v8h*)&lA[r * LDST + kc] =
          *(const v8h*)&A[(size_t)(m0 + r) * lda + kb + kc];
    }
    // ---- stage B tile transposed: lB[n][k]
    if (MODE == 1) {
      // NN: lB[n][k] = B[kb+k][n0+n]  (transpose on the way in)
#pragma unroll
      for (int it = 0; it < 2; ++it) {
        int c  = tid + it * 256;   // 32 k-rows x 16 n-chunks
        int kk = c >> 4;
        int nc = (c & 15) * 8;
        v8h v = *(const v8h*)&B[(size_t)(kb + kk) * ldb + n0 + nc];
#pragma unroll
        for (int e = 0; e < 8; ++e) lB[(nc + e) * LDST + kk] = v[e];
      }
    } else {
      // NT: lB[n][k] = B[n0+n][kb+k]  (rows copied directly)
#pragma unroll
      for (int it = 0; it < 2; ++it) {
        int c  = tid + it * 256;
        int r  = c >> 2;
        int kc = (c & 3) * 8;
        *(v8h*)&lB[r * LDST + kc] =
            *(const v8h*)&B[(size_t)(n0 + r) * ldb + kb + kc];
      }
    }
    __syncthreads();

    // ---- fragments (ISA 16-bit A 16x32 / B 32x16 lane layouts)
    F16x16 afrag[4], bfrag[2];
#pragma unroll
    for (int i = 0; i < 4; ++i) {
      int r   = wm * 64 + i * 16 + row_a;
      int kb2 = hi * 8;
      afrag[i].h[0] = *(const v8h*)&lA[r * LDST + kb2];
      afrag[i].h[1] = *(const v8h*)&lA[r * LDST + kb2 + 16];
    }
#pragma unroll
    for (int j = 0; j < 2; ++j) {
      int cc = wn * 32 + j * 16 + row_a;
      int ks = hi * 8;
      bfrag[j].h[0] = *(const v8h*)&lB[cc * LDST + ks];
      bfrag[j].h[1] = *(const v8h*)&lB[cc * LDST + ks + 16];
    }
#pragma unroll
    for (int i = 0; i < 4; ++i)
#pragma unroll
      for (int j = 0; j < 2; ++j)
        acc[i][j] = __builtin_amdgcn_wmma_f32_16x16x32_f16(
            false, afrag[i].v, false, bfrag[j].v, (short)0, acc[i][j],
            false, false);
  }

  // ---- epilogue
  float sb2 = 0.f, sw2 = 0.f, sata = 0.f;
  if (MODE == 2) {
    float sb = softplusf(raw_sb[0]);
    float sw = softplusf(raw_sw[0]);
    sb2  = sb * sb;
    sw2  = sw * sw;
    sata = softplusf(raw_sa[0]) * softplusf(raw_ta[0]);
  }
#pragma unroll
  for (int i = 0; i < 4; ++i) {
#pragma unroll
    for (int j = 0; j < 2; ++j) {
      int trow = m0 + wm * 64 + i * 16 + hi * 8;
      int tcol = n0 + wn * 32 + j * 16 + (lane & 15);
#pragma unroll
      for (int r = 0; r < 8; ++r) {
        int row = trow + r;
        float v = acc[i][j][r];
        size_t idx = (size_t)row * ldc + tcol;
        if (MODE == 0) {
          float sq = norms[row] + norms[tcol] - 2.0f * v;
          sq = fmaxf(sq, 0.0f);
          ((_Float16*)Cout)[idx] = (_Float16)__expf(-0.5f * sq);
        } else if (MODE == 1) {
          ((_Float16*)Cout)[idx] = (_Float16)v;
        } else {
          float o = (sb2 + sw2 * v) * sata;
          float* O = (float*)Cout;
          O[idx] = accumulate ? (O[idx] + o) : o;
        }
      }
    }
  }
}

// ---------------------------------------------------------------------------
// relu_cov pieces
// ---------------------------------------------------------------------------
__global__ void diag_kernel(const float* __restrict__ OUT,
                            const float* __restrict__ raw_vw,
                            float* __restrict__ svec) {
  int i = blockIdx.x * 256 + threadIdx.x;
  if (i < NN_) {
    float vw = softplusf(raw_vw[0]);
    svec[i] = sqrtf(vw * vw * OUT[(size_t)i * NN_ + i]) + 1e-5f;
  }
}

__global__ void relu_cov_kernel(const float* __restrict__ OUT,
                                const float* __restrict__ svec,
                                const float* __restrict__ raw_vw,
                                _Float16* __restrict__ Kh) {
  size_t idx = (size_t)blockIdx.x * 256 + threadIdx.x;
  int i = (int)(idx / NN_);
  int j = (int)(idx % NN_);
  float vw = softplusf(raw_vw[0]);
  float K  = vw * vw * OUT[idx];
  float ss = svec[i] * svec[j];
  float c  = K / ss;
  c = fminf(fmaxf(c, -1.0f + 1e-6f), 1.0f - 1e-6f);
  float th = acosf(c);
  float r  = 0.15915494309189535f *
             (sinf(th) + (3.14159265358979f - th) * cosf(th)) * ss;
  Kh[idx] = (_Float16)r;
}

// ---------------------------------------------------------------------------
// block means (deterministic two-stage reduction)
// ---------------------------------------------------------------------------
__global__ void block_sum_kernel(const float* __restrict__ OUT,
                                 const float* __restrict__ raw_vw,
                                 float* __restrict__ partial) {
  __shared__ float red[256];
  size_t idx = (size_t)blockIdx.x * 256 + threadIdx.x;
  float vw = softplusf(raw_vw[0]);
  red[threadIdx.x] = vw * vw * OUT[idx];
  __syncthreads();
  for (int s = 128; s > 0; s >>= 1) {
    if (threadIdx.x < s) red[threadIdx.x] += red[threadIdx.x + s];
    __syncthreads();
  }
  if (threadIdx.x == 0) partial[blockIdx.x] = red[0];
}

__global__ void reduce_partials_kernel(const float* __restrict__ partial,
                                       float* __restrict__ means) {
  __shared__ float red[3][256];
  float loc[3] = {0.f, 0.f, 0.f};
  const int nblk = (NN_ * (NN_ / 256));
  for (int b = threadIdx.x; b < nblk; b += 256) {
    int i  = b >> 4;          // row (4096/256 = 16 blocks per row)
    int j0 = (b & 15) * 256;  // first col of block (region-uniform)
    int which = (i < SS_) ? (j0 < SS_ ? 0 : 2) : (j0 < SS_ ? 3 : 1);
    if (which != 3) loc[which] += partial[b];
  }
  for (int r = 0; r < 3; ++r) red[r][threadIdx.x] = loc[r];
  __syncthreads();
  for (int s = 128; s > 0; s >>= 1) {
    if (threadIdx.x < s)
      for (int r = 0; r < 3; ++r)
        red[r][threadIdx.x] += red[r][threadIdx.x + s];
    __syncthreads();
  }
  if (threadIdx.x < 3)
    means[threadIdx.x] = red[threadIdx.x][0] / (float)((size_t)SS_ * SS_);
}

// ---------------------------------------------------------------------------
// final gather + index_K scaling
// ---------------------------------------------------------------------------
__global__ void final_kernel(const float* __restrict__ OUT,
                             const float* __restrict__ raw_vw,
                             const float* __restrict__ means,
                             const int* __restrict__ x1,
                             const int* __restrict__ x2,
                             float* __restrict__ out) {
  size_t idx = (size_t)blockIdx.x * 256 + threadIdx.x;
  int i = (int)(idx / NN_);
  int j = (int)(idx % NN_);
  int ii = x1[i];
  int jj = x2[j];
  float vw = softplusf(raw_vw[0]);
  float m;
  if (ii < SS_ && jj < SS_)        m = means[0];
  else if (ii >= SS_ && jj >= SS_) m = means[1];
  else                             m = means[2];
  out[idx] = vw * vw * OUT[(size_t)ii * NN_ + jj] * m;
}

// ---------------------------------------------------------------------------
// launch
// ---------------------------------------------------------------------------
extern "C" void kernel_launch(void* const* d_in, const int* in_sizes, int n_in,
                              void* d_out, int out_size, void* d_ws,
                              size_t ws_size, hipStream_t stream) {
  (void)in_sizes; (void)n_in; (void)out_size; (void)ws_size;
  const float* feat    = (const float*)d_in[0];
  const float* adj     = (const float*)d_in[1];
  const float* raw_ls  = (const float*)d_in[2];
  const float* raw_vw  = (const float*)d_in[3];
  const float* raw_sw  = (const float*)d_in[4];
  const float* raw_sb  = (const float*)d_in[5];
  const float* raw_sa  = (const float*)d_in[6];  // [2]
  const float* raw_ta  = (const float*)d_in[7];  // [2]
  const int*   x1      = (const int*)d_in[8];
  const int*   x2      = (const int*)d_in[9];
  float*       out     = (float*)d_out;

  const size_t NN2 = (size_t)NN_ * NN_;

  char* w = (char*)d_ws;
  auto alloc = [&](size_t bytes) -> void* {
    void* p = (void*)w;
    w += (bytes + 255) & ~(size_t)255;
    return p;
  };
  _Float16* xh      = (_Float16*)alloc((size_t)NN_ * DD_ * 2);
  float*    norms   = (float*)alloc((size_t)NN_ * 4);
  float*    svec    = (float*)alloc((size_t)NN_ * 4);
  float*    means   = (float*)alloc(256);
  float*    partial = (float*)alloc((NN2 / 256) * 4);
  _Float16* Ah      = (_Float16*)alloc(2 * NN2 * 2);
  _Float16* Kh      = (_Float16*)alloc(NN2 * 2);
  _Float16* Bh      = (_Float16*)alloc(NN2 * 2);
  float*    OUT     = (float*)alloc(NN2 * 4);

  dim3 ggrid(NN_ / BN, NN_ / BM);
  int  nblk256 = (int)(NN2 / 256);

  // features + adjacency prep
  prep_x_kernel<<<NN_, 128, 0, stream>>>(feat, raw_ls, xh, norms);
  pack_adj_kernel<<<(int)(2 * NN2 / 256), 256, 0, stream>>>(adj, Ah);

  // C0 = exp(-0.5 * sqdist)  (NT GEMM over D=128 with RBF epilogue) -> Kh f16
  gemm_kernel<0><<<ggrid, 256, 0, stream>>>(xh, xh, Kh, norms, nullptr,
                                            nullptr, nullptr, nullptr, 0,
                                            DD_, DD_, DD_, NN_);

  for (int layer = 0; layer < 2; ++layer) {
    if (layer == 1) {
      // Kh = relu_cov(vw^2 * OUT)
      diag_kernel<<<NN_ / 256, 256, 0, stream>>>(OUT, raw_vw, svec);
      relu_cov_kernel<<<nblk256, 256, 0, stream>>>(OUT, svec, raw_vw, Kh);
    }
    for (int a = 0; a < 2; ++a) {
      const _Float16* Aa = Ah + (size_t)a * NN2;
      // Bh = f16(A @ Kh)           (NN GEMM)
      gemm_kernel<1><<<ggrid, 256, 0, stream>>>(Aa, Kh, Bh, nullptr, nullptr,
                                                nullptr, nullptr, nullptr, 0,
                                                NN_, NN_, NN_, NN_);
      // OUT (+)= (sb^2 + sw^2 * (A @ Bh^T)) * sa*ta   (NT GEMM)
      gemm_kernel<2><<<ggrid, 256, 0, stream>>>(Aa, Bh, OUT, nullptr, raw_sb,
                                                raw_sw, raw_sa + a, raw_ta + a,
                                                a, NN_, NN_, NN_, NN_);
    }
  }

  // block means of K_final = vw^2 * OUT (deterministic reduction)
  block_sum_kernel<<<nblk256, 256, 0, stream>>>(OUT, raw_vw, partial);
  reduce_partials_kernel<<<1, 256, 0, stream>>>(partial, means);

  // out[i,j] = K_final[x1[i], x2[j]] * index_K
  final_kernel<<<nblk256, 256, 0, stream>>>(OUT, raw_vw, means, x1, x2, out);
}
